// HungarianMatcher_52733608460634
// MI455X (gfx1250) — matrix operand order repaired and checked by reading, not compiled, not run on hardware
//
#include <hip/hip_runtime.h>
#include <hip/hip_bf16.h>

typedef __attribute__((ext_vector_type(16))) _Float16     v16h;
typedef __attribute__((ext_vector_type(8)))  float        v8f;
typedef __attribute__((ext_vector_type(8)))  unsigned int v8u;

#define B_   64
#define Q_   1000
#define QP_  1008      // Q padded to multiple of 16
#define C_   256
#define G_   128
#define QT_  63        // QP_/16
#define TPB  256

// ---------------------------------------------------------------------------
// Kernel 1: row softmax (f32 in) -> f16 probabilities, zero-padded to QP_ rows
// One wave (32 lanes) per row; 8 rows per 256-thread block.
// ---------------------------------------------------------------------------
__global__ void __launch_bounds__(256)
softmax_f16_kernel(const float* __restrict__ logits, _Float16* __restrict__ prob)
{
    int tid  = threadIdx.x;
    int wave = tid >> 5;
    int lane = tid & 31;
    int row  = blockIdx.x * 8 + wave;          // [0, B_*QP_)
    if (row >= B_ * QP_) return;
    int b  = row / QP_;
    int qp = row - b * QP_;
    _Float16* orow = prob + ((size_t)(b * QP_ + qp)) * C_;
    if (qp >= Q_) {
        #pragma unroll
        for (int i = 0; i < 8; ++i) orow[lane + 32 * i] = (_Float16)0.0f;
        return;
    }
    const float* irow = logits + ((size_t)(b * Q_ + qp)) * C_;
    float vals[8];
    float m = -3.0e38f;
    #pragma unroll
    for (int i = 0; i < 8; ++i) {
        float x = irow[lane + 32 * i];
        vals[i] = x;
        m = fmaxf(m, x);
    }
    #pragma unroll
    for (int off = 16; off > 0; off >>= 1)
        m = fmaxf(m, __shfl_xor(m, off, 32));
    float s = 0.0f;
    #pragma unroll
    for (int i = 0; i < 8; ++i) {
        float e = __expf(vals[i] - m);
        vals[i] = e;
        s += e;
    }
    #pragma unroll
    for (int off = 16; off > 0; off >>= 1)
        s += __shfl_xor(s, off, 32);
    float inv = 1.0f / s;
    #pragma unroll
    for (int i = 0; i < 8; ++i)
        orow[lane + 32 * i] = (_Float16)(vals[i] * inv);
}

// ---------------------------------------------------------------------------
// Kernel 2: cost matrix via WMMA one-hot gather GEMM + fused bbox/GIoU epilogue
// Block = 256 threads = 8 waves; blockIdx.x = q-tile (16 rows), blockIdx.y = b.
// Wave w computes the 16x16 tile for g-tile w over K = C_ = 256 (8 WMMA steps).
// One-hot B fragments are synthesized with one bucket compare + 8 pair-wise
// 32-bit selects per ktile (exactly one K position is hot per lane).
// ---------------------------------------------------------------------------
__global__ void __launch_bounds__(256)
cost_matrix_kernel(const _Float16* __restrict__ prob,
                   const float*    __restrict__ pred_boxes,
                   const int*      __restrict__ tgt_labels,
                   const float*    __restrict__ tgt_boxes,
                   float*          __restrict__ Cmat)
{
    const int b   = blockIdx.y;
    const int qt  = blockIdx.x;         // 0..62
    const int tid = threadIdx.x;
    const int wave = tid >> 5;          // g-tile index 0..7
    const int lane = tid & 31;

    __shared__ float4 s_pb[16];
    __shared__ float4 s_tb[G_];

    if (tid < G_) {
        const float* t = tgt_boxes + ((size_t)(b * G_ + tid)) * 4;
        s_tb[tid] = make_float4(t[0], t[1], t[2], t[3]);
    }
    if (tid < 16) {
        int q = qt * 16 + tid;
        if (q < Q_) {
            const float* p = pred_boxes + ((size_t)(b * Q_ + q)) * 4;
            s_pb[tid] = make_float4(p[0], p[1], p[2], p[3]);
        } else {
            s_pb[tid] = make_float4(0.f, 0.f, 0.f, 0.f);
        }
    }
    __syncthreads();

    const int h   = lane >> 4;          // lane half
    const int r16 = lane & 15;
    const int g   = wave * 16 + r16;    // N index (B fragment / D column)
    const int label = tgt_labels[b * G_ + g];

    // One-hot decomposition: k = 32*kt + 16*h + e  hot iff (label>>4)==2*kt+h,
    // with e = label&15.  Pre-shift the f16 1.0 pattern into its 32-bit pair.
    const int          bucket  = label >> 4;          // which (kt,h) 16-group
    const int          pairIdx = (label >> 1) & 7;    // 32-bit pair within frag
    const unsigned int oneBits = 0x3C00u << ((label & 1) * 16);

    const _Float16* aRow = prob + ((size_t)(b * QP_ + qt * 16 + r16)) * C_;

    v8f acc = {};
    #pragma unroll
    for (int kt = 0; kt < 8; ++kt) {
        const int kbase = kt * 32;
        // A fragment: contiguous k order, k = 16*h + s (matches one-hot B)
        v16h a;
        #pragma unroll
        for (int v = 0; v < 4; ++v) {
            int k0 = kbase + h * 16 + 2 * v;
            a[2 * v]     = aRow[k0];
            a[2 * v + 1] = aRow[k0 + 1];
        }
        #pragma unroll
        for (int v = 4; v < 8; ++v) {
            int k0 = kbase + h * 16 + 8 + 2 * (v - 4);
            a[2 * v]     = aRow[k0];
            a[2 * v + 1] = aRow[k0 + 1];
        }
        // B fragment: one-hot(labels), built as 8 32-bit pair selects
        const int hv = (bucket == 2 * kt + h) ? pairIdx : 8;   // 8 = never
        v8u bu;
        #pragma unroll
        for (int v = 0; v < 8; ++v)
            bu[v] = (hv == v) ? oneBits : 0u;
        v16h bf = __builtin_bit_cast(v16h, bu);

        if (kt < 7) __builtin_prefetch(aRow + kbase + 32, 0, 3);
        acc = __builtin_amdgcn_wmma_f32_16x16x32_f16(
                  false, a, false, bf, (short)0, acc, false, false);
    }

    // Target box terms (fixed per lane column)
    const float4 tb = s_tb[g];
    const float tx0 = tb.x - 0.5f * tb.z, ty0 = tb.y - 0.5f * tb.w;
    const float tx1 = tb.x + 0.5f * tb.z, ty1 = tb.y + 0.5f * tb.w;
    const float a2  = (tx1 - tx0) * (ty1 - ty0);

    float* outB = Cmat + (size_t)b * Q_ * G_;
    #pragma unroll
    for (int r = 0; r < 8; ++r) {
        const int mrow = r + 8 * h;     // D layout: M = r + 8*(lane>>4)
        const int q = qt * 16 + mrow;
        const float4 pb = s_pb[mrow];
        const float px0 = pb.x - 0.5f * pb.z, py0 = pb.y - 0.5f * pb.w;
        const float px1 = pb.x + 0.5f * pb.z, py1 = pb.y + 0.5f * pb.w;
        const float a1 = (px1 - px0) * (py1 - py0);
        const float ltx = fmaxf(px0, tx0), lty = fmaxf(py0, ty0);
        const float rbx = fminf(px1, tx1), rby = fminf(py1, ty1);
        const float iw = fmaxf(rbx - ltx, 0.f), ih = fmaxf(rby - lty, 0.f);
        const float inter = iw * ih;
        const float sa = a1 + a2;
        const float iou = inter / (sa - inter);
        const float lEx = fminf(px0, tx0), lEy = fminf(py0, ty0);
        const float rEx = fmaxf(px1, tx1), rEy = fmaxf(py1, ty1);
        const float ew = fmaxf(rEx - lEx, 0.f), eh = fmaxf(rEy - lEy, 0.f);
        const float enc = ew * eh;
        // faithfully replicates the reference's iou*enc variant
        const float giou = iou - (enc - (sa - iou * enc)) / enc;
        const float cbbox = fabsf(pb.x - tb.x) + fabsf(pb.y - tb.y)
                          + fabsf(pb.z - tb.z) + fabsf(pb.w - tb.w);
        const float val = -acc[r] + 5.0f * cbbox - 2.0f * giou;
        if (q < Q_) outB[(size_t)q * G_ + g] = val;
    }
}

// ---------------------------------------------------------------------------
// Kernel 3: Jonker-Volgenant assignment per batch (n=G_ rows, m=Q_ cols on C^T)
// One workgroup per batch. Fused minv-update + local argmin (each thread owns
// a disjoint column set), wave32 shfl (value,index) reduction, 3 barriers per
// augmenting step. Tie-breaks match np.argmin (min value, then min index).
// ---------------------------------------------------------------------------
__global__ void __launch_bounds__(TPB)
hungarian_kernel(const float* __restrict__ Cmat,
                 float* __restrict__ rows_out,
                 float* __restrict__ cols_out)
{
    const int b   = blockIdx.x;
    const int tid = threadIdx.x;
    const int lane = tid & 31;
    const int wv   = tid >> 5;
    const int m = Q_;   // columns (queries)
    const int n = G_;   // rows (targets)

    __shared__ float sv[Q_ + 1];
    __shared__ float sminv[Q_ + 1];
    __shared__ int   sway[Q_ + 1];
    __shared__ int   sp[Q_ + 1];
    __shared__ unsigned char sused[Q_ + 1];
    __shared__ float su[G_ + 1];
    __shared__ float rval[TPB / 32];
    __shared__ int   ridx[TPB / 32];
    __shared__ int   s_j0;
    __shared__ float s_delta;
    __shared__ int   s_done;

    for (int j = tid; j <= m; j += TPB) { sv[j] = 0.f; sp[j] = 0; }
    for (int i = tid; i <= n; i += TPB) su[i] = 0.f;
    __syncthreads();

    const float* costB = Cmat + (size_t)b * Q_ * G_;   // cost(i,j)=costB[j*G_+i]
    const float INF = 1.0e30f;

    for (int i = 1; i <= n; ++i) {
        if (tid == 0) { sp[0] = i; s_j0 = 0; }
        for (int j = tid; j <= m; j += TPB) { sminv[j] = INF; sused[j] = 0; }
        __syncthreads();
        while (true) {
            const int   j0  = s_j0;
            const int   i0  = sp[j0];
            const float ui0 = su[i0];
            // mark j0 used (owning thread; j0==0 handled by tid 0)
            if (j0 == 0) { if (tid == 0) sused[0] = 1; }
            else if (tid == ((j0 - 1) & (TPB - 1))) sused[j0] = 1;

            // fused: update minv/way over owned free columns + local argmin
            float best = INF; int bidx = m + 1;
            for (int j = tid + 1; j <= m; j += TPB) {
                if (j == j0 || sused[j]) continue;
                float mv  = sminv[j];
                float cur = costB[(size_t)(j - 1) * G_ + (i0 - 1)] - ui0 - sv[j];
                if (cur < mv) { mv = cur; sminv[j] = cur; sway[j] = j0; }
                if (mv < best || (mv == best && j < bidx)) { best = mv; bidx = j; }
            }
            // wave32 (value,index) min-reduction
            #pragma unroll
            for (int off = 16; off > 0; off >>= 1) {
                float ov = __shfl_xor(best, off, 32);
                int   oi = __shfl_xor(bidx, off, 32);
                if (ov < best || (ov == best && oi < bidx)) { best = ov; bidx = oi; }
            }
            if (lane == 0) { rval[wv] = best; ridx[wv] = bidx; }
            __syncthreads();                                   // barrier A
            if (tid == 0) {
                float bv = rval[0]; int bi = ridx[0];
                #pragma unroll
                for (int w = 1; w < TPB / 32; ++w) {
                    float ov = rval[w]; int oi = ridx[w];
                    if (ov < bv || (ov == bv && oi < bi)) { bv = ov; bi = oi; }
                }
                s_delta = bv;
                s_j0    = bi;
                s_done  = (sp[bi] == 0) ? 1 : 0;   // p untouched by updates
            }
            __syncthreads();                                   // barrier B
            const float delta = s_delta;
            for (int j = tid; j <= m; j += TPB) {
                if (sused[j]) { su[sp[j]] += delta; sv[j] -= delta; }
                else          { sminv[j] -= delta; }
            }
            __syncthreads();                                   // barrier C
            if (s_done) break;
        }
        if (tid == 0) {                      // augment along alternating path
            int j0 = s_j0;
            while (j0 != 0) { int j1 = sway[j0]; sp[j0] = sp[j1]; j0 = j1; }
        }
        // next row's init barrier orders sp updates before the next scan
    }
    if (tid == 0) {                          // emit (query, gt) sorted by query
        int k = 0;
        for (int j = 1; j <= m; ++j) {
            if (sp[j] != 0) {
                rows_out[b * G_ + k] = (float)(j - 1);
                cols_out[b * G_ + k] = (float)(sp[j] - 1);
                ++k;
            }
        }
    }
}

// ---------------------------------------------------------------------------
extern "C" void kernel_launch(void* const* d_in, const int* in_sizes, int n_in,
                              void* d_out, int out_size, void* d_ws, size_t ws_size,
                              hipStream_t stream)
{
    (void)in_sizes; (void)n_in; (void)out_size; (void)ws_size;
    const float* pred_logits = (const float*)d_in[0];   // [B,Q,C] f32
    const float* pred_boxes  = (const float*)d_in[1];   // [B,Q,4] f32
    const int*   tgt_labels  = (const int*)  d_in[2];   // [B,G]   i32
    const float* tgt_boxes   = (const float*)d_in[3];   // [B,G,4] f32

    float* Cmat     = (float*)d_out;                        // [B,Q,G]
    float* rows_out = Cmat + (size_t)B_ * Q_ * G_;          // [B,G]
    float* cols_out = rows_out + (size_t)B_ * G_;           // [B,G]

    _Float16* prob = (_Float16*)d_ws;                       // [B,QP_,C] f16

    const int rowsTotal = B_ * QP_;
    softmax_f16_kernel<<<(rowsTotal + 7) / 8, 256, 0, stream>>>(pred_logits, prob);

    dim3 gridCost(QT_, B_);
    cost_matrix_kernel<<<gridCost, 256, 0, stream>>>(prob, pred_boxes,
                                                     tgt_labels, tgt_boxes, Cmat);

    hungarian_kernel<<<B_, TPB, 0, stream>>>(Cmat, rows_out, cols_out);
}
